// Bert_Seq_5738076307915
// MI455X (gfx1250) — matrix operand order repaired and checked by reading, not compiled, not run on hardware
//
#include <hip/hip_runtime.h>
#include <hip/hip_bf16.h>

// CRF log-likelihood for S=512, B=512, T=48.
//   numerator: gather-based score (thread per batch)
//   normalizer: forward algorithm, logsumexp-matmul rewritten as
//               P@E matmul on v_wmma_f32_16x16x32_f16 (E = exp(transitions),
//               constant B-fragments held in VGPRs for the whole scan)

#define S_LEN 512
#define B_SZ  512
#define T_TAG 48

typedef __attribute__((ext_vector_type(16))) _Float16 v16h;
typedef __attribute__((ext_vector_type(8)))  _Float16 v8h;
typedef __attribute__((ext_vector_type(8)))  float    v8f;

#define WAVES_PER_BLOCK 4
#define ROWS_PER_WAVE   16      // WMMA M tile
#define PSTR 72                 // halves per P row (144 B: 16B-aligned, bank-spread)
#define SSTR 56                 // floats per score row (224 B: 16B-aligned)

// Same-wave LDS producer->consumer ordering: DS ops from one wave are in-order,
// lanes are lockstep; we only need the counter drained + a compiler fence.
__device__ __forceinline__ void lds_fence() {
    asm volatile("s_wait_dscnt 0" ::: "memory");
}

__global__ void zero_out_kernel(float* out) { out[0] = 0.0f; }

// ---------------- numerator: one thread per batch element ----------------
__global__ void crf_numerator_kernel(const float* __restrict__ emis,
                                     const int* __restrict__ tags,
                                     const unsigned char* __restrict__ mask,
                                     const float* __restrict__ start_t,
                                     const float* __restrict__ end_t,
                                     const float* __restrict__ trans,
                                     float* __restrict__ out)
{
    int b = blockIdx.x * blockDim.x + threadIdx.x;
    if (b >= B_SZ) return;

    int prev = tags[b];                                     // tags[0, b]
    float sc = start_t[prev] + emis[(size_t)b * T_TAG + prev];
    int cnt = 1;
    for (int t = 1; t < S_LEN; ++t) {
        int tg   = tags[t * B_SZ + b];
        bool mk  = mask[t * B_SZ + b] != 0;
        float mf = mk ? 1.0f : 0.0f;
        sc += mf * (trans[prev * T_TAG + tg] +
                    emis[((size_t)t * B_SZ + b) * T_TAG + tg]);
        cnt += mk ? 1 : 0;
        prev = tg;
    }
    int last = tags[(cnt - 1) * B_SZ + b];
    sc += end_t[last];
    atomicAdd(out, sc);
}

// ---------------- normalizer: forward algorithm on WMMA ----------------
// One wave per 16 batch rows. Score (16 x 48 f32) in LDS. Per step:
//   rowmax -> P = exp(score - rowmax) as f16 (K padded 48->64 with zeros)
//   -> 2 A-fragments from LDS -> 6 WMMAs against constant E fragments
//   -> score = mask ? rowmax + log(acc) + emis : score
__global__ __launch_bounds__(WAVES_PER_BLOCK * 32)
void crf_normalizer_kernel(const float* __restrict__ emis,
                           const unsigned char* __restrict__ mask,
                           const float* __restrict__ start_t,
                           const float* __restrict__ end_t,
                           const float* __restrict__ trans,
                           float* __restrict__ out)
{
    __shared__ __align__(16) float s_score[WAVES_PER_BLOCK][ROWS_PER_WAVE][SSTR];
    __shared__ __align__(16) _Float16 s_p[WAVES_PER_BLOCK][ROWS_PER_WAVE][PSTR];
    __shared__ float s_rmax[WAVES_PER_BLOCK][ROWS_PER_WAVE];

    const int lane  = threadIdx.x & 31;
    const int w     = threadIdx.x >> 5;
    const int bbase = (blockIdx.x * WAVES_PER_BLOCK + w) * ROWS_PER_WAVE;

    const int r  = lane & 15;   // row this lane owns for rowmax / P production
    const int h  = lane >> 4;   // half-wave select
    const int j0 = h * 24;      // this lane covers j in [j0, j0+24)

    // --- constant B fragments: E[i][j] = exp(trans[i][j]); 3 N-tiles x 2 K-chunks.
    // B (32x16), contiguous k order shared with A: lane = N (mod 16), half e -> K = 16*h + e (+32*kchunk).
    v16h Bf[3][2];
    for (int nt = 0; nt < 3; ++nt) {
        for (int kc = 0; kc < 2; ++kc) {
            v16h bv;
            int N = nt * 16 + r;
            for (int e = 0; e < 16; ++e) {
                int K = kc * 32 + h * 16 + e;
                float v = (K < T_TAG) ? __expf(trans[K * T_TAG + N]) : 0.0f;
                bv[e] = (_Float16)v;
            }
            Bf[nt][kc] = bv;
        }
    }

    // --- init: score0 = start_transitions + emissions[0]; zero P pad K=48..63
    for (int jj = 0; jj < 24; ++jj) {
        int j = j0 + jj;
        s_score[w][r][j] = start_t[j] + emis[(size_t)(bbase + r) * T_TAG + j];
    }
    for (int jj = 0; jj < 8; ++jj)
        s_p[w][r][48 + h * 8 + jj] = (_Float16)0.0f;
    lds_fence();

    for (int t = 1; t < S_LEN; ++t) {
        // read this lane's 24 score values once (vectorized b128 LDS loads)
        float sv[24];
        {
            const float4* srow = (const float4*)&s_score[w][r][j0];
#pragma unroll
            for (int q = 0; q < 6; ++q) {
                float4 x = srow[q];
                sv[q * 4 + 0] = x.x; sv[q * 4 + 1] = x.y;
                sv[q * 4 + 2] = x.z; sv[q * 4 + 3] = x.w;
            }
        }
        // rowmax over 48 tags (24/24 across half-waves, combine via shfl)
        float mx = sv[0];
#pragma unroll
        for (int jj = 1; jj < 24; ++jj) mx = fmaxf(mx, sv[jj]);
        mx = fmaxf(mx, __shfl_xor(mx, 16, 32));
        if (lane < 16) s_rmax[w][r] = mx;

        // P = exp(score - rowmax) in f16
#pragma unroll
        for (int jj = 0; jj < 24; ++jj)
            s_p[w][r][j0 + jj] = (_Float16)__expf(sv[jj] - mx);
        lds_fence();

        // A fragments (A 16x32, same contiguous k order as B): lane row = r,
        // lo 8 halves K = 16h + [0..7], hi 8 halves K = 16h + 8 + [0..7]
        union { v16h v; v8h p[2]; } A0, A1;
        const int klo = h * 16;
        const int khi = h * 16 + 8;
        A0.p[0] = *(const v8h*)&s_p[w][r][klo];
        A0.p[1] = *(const v8h*)&s_p[w][r][khi];
        A1.p[0] = *(const v8h*)&s_p[w][r][32 + klo];
        A1.p[1] = *(const v8h*)&s_p[w][r][32 + khi];

        union { v8f v; float f[8]; } acc[3];
#pragma unroll
        for (int nt = 0; nt < 3; ++nt) {
            v8f c = {};
            c = __builtin_amdgcn_wmma_f32_16x16x32_f16(false, A0.v, false, Bf[nt][0],
                                                       (short)0, c, false, false);
            c = __builtin_amdgcn_wmma_f32_16x16x32_f16(false, A1.v, false, Bf[nt][1],
                                                       (short)0, c, false, false);
            acc[nt].v = c;
        }

        if (t + 1 < S_LEN)  // warm caches for next step's emissions tile
            __builtin_prefetch(&emis[((size_t)(t + 1) * B_SZ + bbase + r) * T_TAG], 0, 3);

        // per-row constants hoisted out of the nt loop
        float rmx[8];
        bool  mk[8];
#pragma unroll
        for (int v = 0; v < 8; ++v) {
            int m  = v + h * 8;
            rmx[v] = s_rmax[w][m];
            mk[v]  = mask[t * B_SZ + bbase + m] != 0;
        }

        // write-back in C/D layout: vgpr v -> M = v + 8h, lane -> N = nt*16 + r
#pragma unroll
        for (int nt = 0; nt < 3; ++nt) {
            int j = nt * 16 + r;
#pragma unroll
            for (int v = 0; v < 8; ++v) {
                int m = v + h * 8;
                float e  = emis[((size_t)t * B_SZ + bbase + m) * T_TAG + j];
                float ns = rmx[v] + __logf(acc[nt].f[v]) + e;
                float os = s_score[w][m][j];
                s_score[w][m][j] = mk[v] ? ns : os;
            }
        }
        lds_fence();
    }

    // denominator: logsumexp_j(score + end_transitions)
    float mx = -3.0e38f;
    for (int jj = 0; jj < 24; ++jj)
        mx = fmaxf(mx, s_score[w][r][j0 + jj] + end_t[j0 + jj]);
    mx = fmaxf(mx, __shfl_xor(mx, 16, 32));
    float sum = 0.0f;
    for (int jj = 0; jj < 24; ++jj)
        sum += __expf(s_score[w][r][j0 + jj] + end_t[j0 + jj] - mx);
    sum += __shfl_xor(sum, 16, 32);
    if (lane < 16) {
        float den = mx + __logf(sum);
        atomicAdd(out, -den);   // llh = numerator - denominator
    }
}

extern "C" void kernel_launch(void* const* d_in, const int* in_sizes, int n_in,
                              void* d_out, int out_size, void* d_ws, size_t ws_size,
                              hipStream_t stream) {
    (void)in_sizes; (void)n_in; (void)out_size; (void)d_ws; (void)ws_size;
    const float*         emis    = (const float*)d_in[0];
    const int*           tags    = (const int*)d_in[1];
    const unsigned char* mask    = (const unsigned char*)d_in[2];
    const float*         start_t = (const float*)d_in[3];
    const float*         end_t   = (const float*)d_in[4];
    const float*         trans   = (const float*)d_in[5];
    float*               out     = (float*)d_out;

    zero_out_kernel<<<1, 1, 0, stream>>>(out);
    crf_numerator_kernel<<<(B_SZ + 255) / 256, 256, 0, stream>>>(
        emis, tags, mask, start_t, end_t, trans, out);
    crf_normalizer_kernel<<<B_SZ / (WAVES_PER_BLOCK * ROWS_PER_WAVE),
                            WAVES_PER_BLOCK * 32, 0, stream>>>(
        emis, mask, start_t, end_t, trans, out);
}
